// _S4DKernel_54219667144855
// MI455X (gfx1250) — matrix operand order repaired and checked by reading, not compiled, not run on hardware
//
#include <hip/hip_runtime.h>

typedef __attribute__((ext_vector_type(16))) _Float16 v16h;
typedef __attribute__((ext_vector_type(8)))  float    v8f;

namespace {

constexpr int NH    = 32;    // N/2
constexpr int L_LEN = 8192;
constexpr int M_DIM = 128;   // t_hi count (L = M_DIM * N_DIM)
constexpr int N_DIM = 64;    // t_lo count
// GEMM per head: M=128, N=64, K=64 (K = 2*NH, real/imag interleaved)

// --- LDS fragment-major index helpers (match CDNA5 WMMA VGPR layouts) ---
// A 16x32 f16 frag: lanes 0-15 hold K{0..7}∪{16..23}, lanes 16-31 K{8..15}∪{24..31}
__device__ __forceinline__ int a_idx(int m, int k) {
    int tile_m = m >> 4, row = m & 15;
    int kc = k >> 5, kk = k & 31;
    int lane = row + (((kk >> 3) & 1) << 4);
    int elem = (((kk >> 4) & 1) << 3) + (kk & 7);
    return (((tile_m << 1) + kc) * 32 + lane) * 16 + elem;
}
// B 32x16 f16 frag: lanes 0-15 hold K{0..7}∪{16..23} (N=lane%16), lanes 16-31 K{8..15}∪{24..31}
__device__ __forceinline__ int b_idx(int k, int n) {
    int tile_n = n >> 4, col = n & 15;
    int kc = k >> 5, kk = k & 31;
    int lane = col + (((kk >> 3) & 1) << 4);
    int elem = (((kk >> 4) & 1) << 3) + (kk & 7);
    return (((tile_n << 1) + kc) * 32 + lane) * 16 + elem;
}

__device__ __forceinline__ void split_f16(float x, _Float16& hi, _Float16& lo) {
    hi = (_Float16)x;
    lo = (_Float16)(x - (float)hi);
}

} // namespace

__global__ __launch_bounds__(256) void s4d_wmma_kernel(
    const float* __restrict__ C_real, const float* __restrict__ C_imag,
    const float* __restrict__ log_dt, const float* __restrict__ log_a_real,
    const float* __restrict__ A_imag, float* __restrict__ out)
{
    // 48 KB LDS: split-f16 A (128x64) and B (64x64), fragment-major
    __shared__ __align__(32) _Float16 sAhi[M_DIM * 64];
    __shared__ __align__(32) _Float16 sAlo[M_DIM * 64];
    __shared__ __align__(32) _Float16 sBhi[64 * N_DIM];
    __shared__ __align__(32) _Float16 sBlo[64 * N_DIM];

    const int h   = blockIdx.x;
    const int tid = threadIdx.x;
    const int n   = tid & (NH - 1);   // each thread owns one state n
    const int sub = tid >> 5;         // 0..7 (also the wave id)

    // ---------------- Phase 1: seeds, C', G=C'*U, V -> LDS ----------------
    const float dt = expf(log_dt[h]);
    const float ar = -expf(log_a_real[h * NH + n]);
    const float ai = A_imag[h * NH + n];
    const float xr = ar * dt;         // Re(dtA)
    const float xi = ai * dt;         // Im(dtA)

    float s, c;
    const float mag0 = expf(xr);
    sincosf(xi, &s, &c);
    const float er  = mag0 * c - 1.0f;          // exp(dtA) - 1
    const float ei  = mag0 * s;
    const float inv = 1.0f / (ar * ar + ai * ai);
    const float Fr  = (er * ar + ei * ai) * inv; // (exp(dtA)-1)/A
    const float Fi  = (ei * ar - er * ai) * inv;
    const float Cr  = C_real[h * NH + n], Ci = C_imag[h * NH + n];
    const float Gr0 = 2.0f * (Cr * Fr - Ci * Fi); // C' with the 2x folded in
    const float Gi0 = 2.0f * (Cr * Fi + Ci * Fr);

    // A rows: G[m] = C' * exp(dtA * 64m), m = t_hi
    #pragma unroll
    for (int j = 0; j < 16; ++j) {
        const int   m  = sub + (j << 3);
        const float tt = (float)(m << 6);         // 64*m
        const float mg = expf(xr * tt);
        sincosf(xi * tt, &s, &c);
        const float ur = mg * c, ui = mg * s;
        const float gr = Gr0 * ur - Gi0 * ui;
        const float gi = Gr0 * ui + Gi0 * ur;
        _Float16 ph, pl;
        const int i0 = a_idx(m, 2 * n), i1 = a_idx(m, 2 * n + 1);
        split_f16(gr, ph, pl); sAhi[i0] = ph; sAlo[i0] = pl;
        split_f16(gi, ph, pl); sAhi[i1] = ph; sAlo[i1] = pl;
    }
    // B cols: V[tl] = exp(dtA * tl), tl = t_lo; store (Vr, -Vi) on K rows (2n, 2n+1)
    #pragma unroll
    for (int j = 0; j < 8; ++j) {
        const int   tl = sub + (j << 3);
        const float tt = (float)tl;
        const float mg = expf(xr * tt);
        sincosf(xi * tt, &s, &c);
        const float vr =  mg * c;
        const float vi = -(mg * s);
        _Float16 ph, pl;
        const int i0 = b_idx(2 * n, tl), i1 = b_idx(2 * n + 1, tl);
        split_f16(vr, ph, pl); sBhi[i0] = ph; sBlo[i0] = pl;
        split_f16(vi, ph, pl); sBhi[i1] = ph; sBlo[i1] = pl;
    }
    __syncthreads();

    // ---------------- Phase 2: per-wave split-f16 WMMA GEMM ----------------
    const int wave   = tid >> 5;
    const int lane   = tid & 31;
    const int tile_m = wave;                       // 8 waves x 8 row-tiles

    const v16h a_h0 = *(const v16h*)&sAhi[((tile_m * 2 + 0) * 32 + lane) * 16];
    const v16h a_h1 = *(const v16h*)&sAhi[((tile_m * 2 + 1) * 32 + lane) * 16];
    const v16h a_l0 = *(const v16h*)&sAlo[((tile_m * 2 + 0) * 32 + lane) * 16];
    const v16h a_l1 = *(const v16h*)&sAlo[((tile_m * 2 + 1) * 32 + lane) * 16];

    const int row0 = tile_m * 16 + ((lane >> 4) << 3); // D layout: M = r + 8*(lane/16)
    const int col  = lane & 15;                        // D layout: N = lane%16
    float* outh = out + (size_t)h * L_LEN;

    #pragma unroll
    for (int tn = 0; tn < 4; ++tn) {
        const v16h b_h0 = *(const v16h*)&sBhi[((tn * 2 + 0) * 32 + lane) * 16];
        const v16h b_h1 = *(const v16h*)&sBhi[((tn * 2 + 1) * 32 + lane) * 16];
        const v16h b_l0 = *(const v16h*)&sBlo[((tn * 2 + 0) * 32 + lane) * 16];
        const v16h b_l1 = *(const v16h*)&sBlo[((tn * 2 + 1) * 32 + lane) * 16];
        v8f acc = {};
        // hi*hi + hi*lo + lo*hi  (Markidis split, K=64 via two K=32 chunks)
        acc = __builtin_amdgcn_wmma_f32_16x16x32_f16(false, a_h0, false, b_h0, (short)0, acc, false, false);
        acc = __builtin_amdgcn_wmma_f32_16x16x32_f16(false, a_h1, false, b_h1, (short)0, acc, false, false);
        acc = __builtin_amdgcn_wmma_f32_16x16x32_f16(false, a_h0, false, b_l0, (short)0, acc, false, false);
        acc = __builtin_amdgcn_wmma_f32_16x16x32_f16(false, a_h1, false, b_l1, (short)0, acc, false, false);
        acc = __builtin_amdgcn_wmma_f32_16x16x32_f16(false, a_l0, false, b_h0, (short)0, acc, false, false);
        acc = __builtin_amdgcn_wmma_f32_16x16x32_f16(false, a_l1, false, b_h1, (short)0, acc, false, false);
        #pragma unroll
        for (int r = 0; r < 8; ++r) {
            // K[h, 64*(row0+r) + tn*16 + col]; lanes contiguous along t_lo
            outh[(size_t)(row0 + r) * N_DIM + tn * 16 + col] = acc[r];
        }
    }
}

extern "C" void kernel_launch(void* const* d_in, const int* in_sizes, int n_in,
                              void* d_out, int out_size, void* d_ws, size_t ws_size,
                              hipStream_t stream) {
    (void)n_in; (void)out_size; (void)d_ws; (void)ws_size;
    const float* C_real     = (const float*)d_in[0];
    const float* C_imag     = (const float*)d_in[1];
    const float* log_dt     = (const float*)d_in[2];
    const float* log_a_real = (const float*)d_in[3];
    const float* A_imag     = (const float*)d_in[4];
    // d_in[5] is L (scalar int) = 8192, compile-time constant here
    float* out = (float*)d_out;
    const int H = in_sizes[2];   // log_dt has H elements (512)

    s4d_wmma_kernel<<<dim3(H), dim3(256), 0, stream>>>(
        C_real, C_imag, log_dt, log_a_real, A_imag, out);
}
